// TDALoss_2714419331928
// MI455X (gfx1250) — matrix operand order repaired and checked by reading, not compiled, not run on hardware
//
#include <hip/hip_runtime.h>
#include <hip/hip_bf16.h>

typedef __attribute__((ext_vector_type(16))) _Float16 v16h;
typedef __attribute__((ext_vector_type(8)))  float    v8f;

#define T_POINTS   2097152
#define N_DGM      256
#define CHUNK      2048          // points per workgroup (sorted segs -> 1-2 segments per block)
#define PTS_PER_WAVE (CHUNK / 8) // 8 waves of 32
#define TILES_PER_WAVE (PTS_PER_WAVE / 16)

// jnp.nan_to_num(x, posinf=0): nan->0, +inf->0, -inf->-FLT_MAX  (branchless)
__device__ __forceinline__ float fixval(float v) {
    float r = v;
    r = (r ==  __builtin_inff()) ? 0.0f : r;
    r = (r == -__builtin_inff()) ? -3.4028234663852886e38f : r;
    r = (v != v) ? 0.0f : r;
    return r;
}

__global__ void __launch_bounds__(256)
tda_zero_kernel(float* p, int n) {
    int i = blockIdx.x * 256 + threadIdx.x;
    if (i < n) p[i] = 0.0f;
}

// One side: fused point-MLPs + WMMA second layer. Sorted segs => run-based
// accumulation through the WMMA C operand; LDS touched only at run ends.
// Segment probes are readfirstlane'd so all run control is SALU + s_cbranch.
__global__ void __launch_bounds__(256)
tda_pool_kernel(const float* __restrict__ pts, const int* __restrict__ seg,
                const float* __restrict__ w1_w, const float* __restrict__ w1_b,
                const float* __restrict__ w2_w, const float* __restrict__ w2_b,
                const float* __restrict__ e1_w, const float* __restrict__ e1_b,
                const float* __restrict__ e2_w,
                float* __restrict__ gsums,   // [N_DGM*32]
                float* __restrict__ gsumw,   // [N_DGM]
                float* __restrict__ gcnt) {  // [N_DGM]
    __shared__ float lsums[N_DGM * 32];
    __shared__ float lsumw[N_DGM];
    __shared__ float lcnt [N_DGM];

    const int tid = threadIdx.x;
    for (int i = tid; i < N_DGM * 32; i += 256) lsums[i] = 0.0f;
    if (tid < N_DGM) { lsumw[tid] = 0.0f; lcnt[tid] = 0.0f; }
    __syncthreads();

    const int lane = tid & 31;
    const int wv   = tid >> 5;
    const int kh   = lane >> 4;     // which K-half this lane provides
    const int mrow = lane & 15;     // A-row (point within tile) / B,D column
    const int blockStart = blockIdx.x * CHUNK;

    const float w2b0 = w2_b[0];
    const float2* __restrict__ pts2 = reinterpret_cast<const float2*>(pts);

    // Per-lane K indices for the 16-bit 16x32 A layout:
    //  j<8  -> k = 8*kh + j ;  j>=8 -> k = 16 + 8*kh + (j-8)
    float e1r0[16], e1r1[16], e1bv[16];
    float w1r0[16], w1r1[16], w1bv[16], w2c[16];
    #pragma unroll
    for (int j = 0; j < 16; ++j) {
        int k = (j < 8) ? (8 * kh + j) : (16 + 8 * kh + (j - 8));
        e1r0[j] = e1_w[k];      e1r1[j] = e1_w[32 + k]; e1bv[j] = e1_b[k];
        w1r0[j] = w1_w[k];      w1r1[j] = w1_w[32 + k]; w1bv[j] = w1_b[k];
        w2c[j]  = w2_w[k];
    }

    // B fragments: e2_w is [K=32][N=32] row-major; tile0 = cols 0..15, tile1 = cols 16..31.
    // 16-bit 32x16 B layout (same K order as A): j<8 -> k = 8*kh + j ; j>=8 -> k = 16 + 8*kh + (j-8), col = lane&15.
    v16h b0, b1;
    #pragma unroll
    for (int j = 0; j < 16; ++j) {
        int k = (j < 8) ? (8 * kh + j) : (16 + 8 * kh + (j - 8));
        b0[j] = (_Float16)e2_w[k * 32 + mrow];
        b1[j] = (_Float16)e2_w[k * 32 + 16 + mrow];
    }

    const int waveBase = blockStart + wv * PTS_PER_WAVE;

    // Per-tile state (filled by computeTile; each tile computed exactly once).
    float w;            // weight-MLP output of this lane's point
    v16h  a;            // A fragment: w * relu(e-hidden), f16
    int   s0u, s15u;    // seg of first/last row of the tile, in SGPRs

    auto computeTile = [&](int t) {
        const int tileBase = waveBase + t * 16;
        const int p = tileBase + mrow;           // lanes L and L+16 share point L&15
        const float2 xy = pts2[p];               // one global_load_b64
        const float x0 = fixval(xy.x);
        const float x1 = fixval(xy.y);
        // Same address in every lane -> value is wave-uniform; pin it scalar so
        // all run control below is SALU (s_cmp/s_cbranch), not exec masking.
        s0u  = __builtin_amdgcn_readfirstlane(seg[tileBase]);
        s15u = __builtin_amdgcn_readfirstlane(seg[tileBase + 15]);

        // weight MLP: each lane does its 16 K-lanes; combine halves across lane^16.
        float part = 0.0f;
        #pragma unroll
        for (int j = 0; j < 16; ++j) {
            float hw = fmaf(x1, w1r1[j], fmaf(x0, w1r0[j], w1bv[j]));
            hw = fmaxf(hw, 0.0f);
            part = fmaf(hw, w2c[j], part);
        }
        w = part + __shfl_xor(part, 16, 32) + w2b0;

        #pragma unroll
        for (int j = 0; j < 16; ++j) {
            float h = fmaf(x1, e1r1[j], fmaf(x0, e1r0[j], e1bv[j]));
            h = fmaxf(h, 0.0f);
            a[j] = (_Float16)(h * w);
        }
    };

    int t = 0;
    computeTile(0);
    while (t < TILES_PER_WAVE) {
        if (s0u != s15u) {
            // Slow path: segment boundary inside tile (~0.2% of tiles).
            const int myseg = seg[waveBase + t * 16 + mrow];
            v8f c0 = {}; v8f c1 = {};
            c0 = __builtin_amdgcn_wmma_f32_16x16x32_f16(false, a, false, b0, (short)0, c0, false, false);
            c1 = __builtin_amdgcn_wmma_f32_16x16x32_f16(false, a, false, b1, (short)0, c1, false, false);
            // D layout: vgpr r, lanes 0-15 -> row r, lanes 16-31 -> row 8+r; col = lane&15 (+16 tile1)
            const int rowbase = (lane < 16) ? 0 : 8;
            #pragma unroll
            for (int r = 0; r < 8; ++r) {
                int sm = __shfl(myseg, rowbase + r, 32);   // seg of D-row's point
                atomicAdd(&lsums[sm * 32 + mrow],      c0[r]);
                atomicAdd(&lsums[sm * 32 + 16 + mrow], c1[r]);
            }
            if (lane < 16) {                                // one lane per point
                atomicAdd(&lsumw[myseg], w);
                atomicAdd(&lcnt [myseg], 1.0f);
            }
            ++t;
            if (t < TILES_PER_WAVE) computeTile(t);
            continue;
        }

        // Fast path: run of uniform tiles in segment `cur`; accumulate via WMMA C.
        const int cur = s0u;
        v8f c0acc = {}; v8f c1acc = {};
        float wacc = 0.0f, cntacc = 0.0f;
        do {
            c0acc = __builtin_amdgcn_wmma_f32_16x16x32_f16(false, a, false, b0, (short)0, c0acc, false, false);
            c1acc = __builtin_amdgcn_wmma_f32_16x16x32_f16(false, a, false, b1, (short)0, c1acc, false, false);
            wacc  += w;
            cntacc += 1.0f;
            ++t;
            if (t >= TILES_PER_WAVE) break;
            computeTile(t);
        } while (s0u == s15u && s0u == cur);

        // Flush run: column sums of accumulated D. Lane L holds rows 0-7 of its
        // column, lane L+16 rows 8-15; both add their partial to the same slot.
        float cs0 = ((c0acc[0] + c0acc[1]) + (c0acc[2] + c0acc[3])) +
                    ((c0acc[4] + c0acc[5]) + (c0acc[6] + c0acc[7]));
        float cs1 = ((c1acc[0] + c1acc[1]) + (c1acc[2] + c1acc[3])) +
                    ((c1acc[4] + c1acc[5]) + (c1acc[6] + c1acc[7]));
        atomicAdd(&lsums[cur * 32 + mrow],      cs0);
        atomicAdd(&lsums[cur * 32 + 16 + mrow], cs1);
        if (lane < 16) {
            atomicAdd(&lsumw[cur], wacc);
            atomicAdd(&lcnt [cur], cntacc);
        }
    }
    __syncthreads();

    // Sorted segs: this block only touched [dmin, dmax] — flush just that range.
    const int dmin = seg[blockStart];
    const int dmax = seg[blockStart + CHUNK - 1];
    const int nseg = dmax - dmin + 1;
    for (int i = tid; i < nseg * 32; i += 256)
        atomicAdd(&gsums[dmin * 32 + i], lsums[dmin * 32 + i]);
    for (int i = tid; i < nseg; i += 256) {
        atomicAdd(&gsumw[dmin + i], lsumw[dmin + i]);
        atomicAdd(&gcnt [dmin + i], lcnt [dmin + i]);
    }
}

// pooled = (sums + sumw*e2_b) / max(cnt,1); loss = mean((pooled_s - pooled_t)^2)
__global__ void __launch_bounds__(256)
tda_finalize_kernel(const float* __restrict__ ws, const float* __restrict__ e2_b,
                    float* __restrict__ out) {
    const float* sums_t = ws;
    const float* sumw_t = ws + 8192;
    const float* cnt_t  = ws + 8448;
    const float* sums_s = ws + 8704;
    const float* sumw_s = ws + 16896;
    const float* cnt_s  = ws + 17152;

    __shared__ float red[256];
    float acc = 0.0f;
    for (int i = threadIdx.x; i < N_DGM * 32; i += 256) {
        int d = i >> 5, n = i & 31;
        float eb = e2_b[n];
        float pt = (sums_t[i] + sumw_t[d] * eb) / fmaxf(cnt_t[d], 1.0f);
        float ps = (sums_s[i] + sumw_s[d] * eb) / fmaxf(cnt_s[d], 1.0f);
        float df = ps - pt;
        acc = fmaf(df, df, acc);
    }
    red[threadIdx.x] = acc;
    __syncthreads();
    for (int s = 128; s > 0; s >>= 1) {
        if (threadIdx.x < s) red[threadIdx.x] += red[threadIdx.x + s];
        __syncthreads();
    }
    if (threadIdx.x == 0) out[0] = red[0] / (float)(N_DGM * 32);
}

extern "C" void kernel_launch(void* const* d_in, const int* in_sizes, int n_in,
                              void* d_out, int out_size, void* d_ws, size_t ws_size,
                              hipStream_t stream) {
    const float* points_t = (const float*)d_in[0];
    const float* points_s = (const float*)d_in[1];
    const float* w1_w = (const float*)d_in[2];
    const float* w1_b = (const float*)d_in[3];
    const float* w2_w = (const float*)d_in[4];
    const float* w2_b = (const float*)d_in[5];
    const float* e1_w = (const float*)d_in[6];
    const float* e1_b = (const float*)d_in[7];
    const float* e2_w = (const float*)d_in[8];
    const float* e2_b = (const float*)d_in[9];
    const int* seg_t  = (const int*)d_in[10];
    const int* seg_s  = (const int*)d_in[11];

    float* ws = (float*)d_ws;
    // ws layout: sums_t[8192] sumw_t[256] cnt_t[256] sums_s[8192] sumw_s[256] cnt_s[256]
    const int WSN = 17408;
    tda_zero_kernel<<<(WSN + 255) / 256, 256, 0, stream>>>(ws, WSN);

    const int blocks = T_POINTS / CHUNK;   // 1024
    tda_pool_kernel<<<blocks, 256, 0, stream>>>(points_t, seg_t,
        w1_w, w1_b, w2_w, w2_b, e1_w, e1_b, e2_w,
        ws, ws + 8192, ws + 8448);
    tda_pool_kernel<<<blocks, 256, 0, stream>>>(points_s, seg_s,
        w1_w, w1_b, w2_w, w2_b, e1_w, e1_b, e2_w,
        ws + 8704, ws + 16896, ws + 17152);

    tda_finalize_kernel<<<1, 256, 0, stream>>>(ws, e2_b, (float*)d_out);
}
